// Attention_49151605736162
// MI455X (gfx1250) — compile-verified, matrix operand order checked by reading, not run on hardware
//
#include <hip/hip_runtime.h>
#include <hip/hip_bf16.h>

typedef _Float16 half_t;
typedef __attribute__((ext_vector_type(16))) _Float16 v16h;
typedef __attribute__((ext_vector_type(8)))  _Float16 h8;
typedef __attribute__((ext_vector_type(8)))  float    v8f;

#define KMAX 512          // K is 512 in every GEMM here
#define LDT  (KMAX + 8)   // padded LDS row stride (halfs): 16B-aligned, bank-spread

// ---------------------------------------------------------------------------
// WMMA fragment build + multiply for one 32-wide K step.
//   A (16x32 f16): lane L holds row M=L%16; halves 0..7 -> K=kb8..kb8+7,
//                  halves 8..15 -> K=16+kb8..+7, kb8 = 8*(L/16).  (ISA 7.12.2)
//   B fragment read from LDS panel Bt[n][k] (transposed, padded stride LDT)
//   as two 16-byte ds loads per lane.
// ---------------------------------------------------------------------------
__device__ __forceinline__ v8f wmma_step(const half_t* __restrict__ ap,
                                         const half_t* __restrict__ bt,
                                         v8f acc) {
  h8 a0 = *(const h8*)(ap);          // global_load_b128
  h8 a1 = *(const h8*)(ap + 16);     // global_load_b128
  h8 b0 = *(const h8*)(bt);          // ds_load_b128
  h8 b1 = *(const h8*)(bt + 16);     // ds_load_b128
  v16h av, bv;
#pragma unroll
  for (int i = 0; i < 8; ++i) {
    av[i] = a0[i]; av[8 + i] = a1[i];
    bv[i] = b0[i]; bv[8 + i] = b1[i];
  }
  return __builtin_amdgcn_wmma_f32_16x16x32_f16(
      false, av, false, bv, (short)0, acc, false, false);
}

// Generic f16 WMMA GEMM: D[M,N] = A[M,K] x B[K,N] (+ bias[N]).
// 256 threads = 8 waves as 4(M) x 2(N); block computes 64 x 32.
// Requires: M % 64 == 0, N % 32 == 0, K == 512.
template <bool OUT_HALF, bool ADD_BIAS>
__global__ void gemm_f16(const half_t* __restrict__ A, const half_t* __restrict__ B,
                         const float* __restrict__ bias,
                         float* __restrict__ Df, half_t* __restrict__ Dh,
                         int M, int N, int K) {
  __shared__ half_t Bt[32 * LDT];                 // ~33 KB
  int tid  = threadIdx.x;
  int lane = tid & 31, wave = tid >> 5;
  int tilesN = N >> 5;
  int bm = blockIdx.x / tilesN;                   // bm-major: neighbors share A (L2)
  int bn = blockIdx.x % tilesN;
  int n0 = bn << 5;
  // Stage B panel [K x 32] transposed into LDS.
  for (int idx = tid; idx < (K << 5); idx += 256) {
    int k = idx >> 5, n = idx & 31;
    Bt[n * LDT + k] = B[(size_t)k * N + n0 + n];
  }
  __syncthreads();
  int wm = wave >> 1, wn = wave & 1;
  int row = bm * 64 + wm * 16 + (lane & 15);
  int kb8 = (lane >> 4) << 3;
  const half_t* arow = A + (size_t)row * K + kb8;
  const half_t* bt   = Bt + (wn * 16 + (lane & 15)) * LDT + kb8;
  v8f acc = {};
  for (int k0 = 0; k0 < K; k0 += 32)
    acc = wmma_step(arow + k0, bt + k0, acc);
  // C/D layout: VGPR j of lane L -> row +8*(L/16)+j, col = L%16  (ISA 7.12.2)
  int orow = bm * 64 + wm * 16 + ((lane >> 4) << 3);
  int col  = n0 + wn * 16 + (lane & 15);
  float bz = ADD_BIAS ? bias[col] : 0.f;
#pragma unroll
  for (int j = 0; j < 8; ++j) {
    float v = acc[j] + bz;
    if (OUT_HALF) Dh[(size_t)(orow + j) * N + col] = (half_t)v;
    else          Df[(size_t)(orow + j) * N + col] = v;
  }
}

// LSTM pre-activation: z[b,n] = ctx@W_lstm[:512] + h@U_lstm + b_lstm
//                               + W_lstm[512 + text[b,t], n]   (one-hot row)
// 128 threads = 4 waves stacked in M; block computes 64 x 16; both weight
// panels (16 cols each) staged transposed in LDS.
__global__ void lstm_z(const half_t* __restrict__ ctx_h, const half_t* __restrict__ Wl_h,
                       const half_t* __restrict__ h_h,   const half_t* __restrict__ U_h,
                       const float* __restrict__ b_lstm, const int* __restrict__ text,
                       float* __restrict__ z, int t) {
  const int N = 2048, K = 512;
  __shared__ half_t Wt[16 * LDT];
  __shared__ half_t Ut[16 * LDT];
  int tid  = threadIdx.x;
  int lane = tid & 31, wave = tid >> 5;
  int tilesN = N >> 4;
  int bm = blockIdx.x / tilesN;
  int bn = blockIdx.x % tilesN;
  int n0 = bn << 4;
  for (int idx = tid; idx < (K << 4); idx += 128) {
    int k = idx >> 4, n = idx & 15;
    Wt[n * LDT + k] = Wl_h[(size_t)k * N + n0 + n];
    Ut[n * LDT + k] = U_h [(size_t)k * N + n0 + n];
  }
  __syncthreads();
  int row = bm * 64 + wave * 16 + (lane & 15);
  int kb8 = (lane >> 4) << 3;
  const half_t* crow = ctx_h + (size_t)row * K + kb8;
  const half_t* hrow = h_h   + (size_t)row * K + kb8;
  const half_t* wt = Wt + (lane & 15) * LDT + kb8;
  const half_t* ut = Ut + (lane & 15) * LDT + kb8;
  v8f acc = {};
  for (int k0 = 0; k0 < K; k0 += 32) {
    acc = wmma_step(crow + k0, wt + k0, acc);
    acc = wmma_step(hrow + k0, ut + k0, acc);
  }
  int orow = bm * 64 + wave * 16 + ((lane >> 4) << 3);
  int col  = n0 + (lane & 15);
  float bz = b_lstm[col];
#pragma unroll
  for (int j = 0; j < 8; ++j) {
    int b = orow + j;
    int ch = text[b * 26 + t];                    // text is [B, T+1]=[128,26]
    float v = acc[j] + bz + (float)Wl_h[(size_t)(512 + ch) * N + col];
    z[(size_t)b * N + col] = v;
  }
}

// Attention step: per-batch-row block. e=tanh(Hproj+hp)@Wscore, softmax over S,
// context = alpha @ x. Hproj/x read as f16 -> 134MB working set fits 192MB L2.
__global__ void attn_step(const half_t* __restrict__ Hproj, const half_t* __restrict__ xh,
                          const float* __restrict__ hp, const float* __restrict__ Wscore,
                          float* __restrict__ ctx, half_t* __restrict__ ctx_h) {
  const int S = 512, H = 512, C = 512;
  int b = blockIdx.x;
  int tid = threadIdx.x;                 // 256 threads = 8 waves
  int lane = tid & 31, wave = tid >> 5;
  __shared__ float hp_s[H], ws_s[H], e_s[S], red[8];
  hp_s[tid] = hp[b * H + tid];           hp_s[tid + 256] = hp[b * H + tid + 256];
  ws_s[tid] = Wscore[tid];               ws_s[tid + 256] = Wscore[tid + 256];
  __syncthreads();
  // Phase 1: one wave per score row s; lane L covers h = 16L..16L+15 (32B loads)
  for (int s = wave; s < S; s += 8) {
    const half_t* hr = Hproj + ((size_t)b * S + s) * H + lane * 16;
    h8 v0 = *(const h8*)(hr);
    h8 v1 = *(const h8*)(hr + 8);
    float p = 0.f;
#pragma unroll
    for (int i = 0; i < 8; ++i) {
      int hh = lane * 16 + i;
      p += ws_s[hh]     * tanhf((float)v0[i] + hp_s[hh]);
      p += ws_s[hh + 8] * tanhf((float)v1[i] + hp_s[hh + 8]);
    }
#pragma unroll
    for (int off = 16; off > 0; off >>= 1) p += __shfl_xor(p, off, 32);
    if (lane == 0) e_s[s] = p;
  }
  __syncthreads();
  // Softmax over S=512 (each thread owns 2 entries)
  float m = fmaxf(e_s[tid], e_s[tid + 256]);
#pragma unroll
  for (int off = 16; off > 0; off >>= 1) m = fmaxf(m, __shfl_xor(m, off, 32));
  if (lane == 0) red[wave] = m;
  __syncthreads();
  float mall = red[0];
#pragma unroll
  for (int i = 1; i < 8; ++i) mall = fmaxf(mall, red[i]);
  __syncthreads();
  float a0 = __expf(e_s[tid] - mall), a1 = __expf(e_s[tid + 256] - mall);
  float sm = a0 + a1;
#pragma unroll
  for (int off = 16; off > 0; off >>= 1) sm += __shfl_xor(sm, off, 32);
  if (lane == 0) red[wave] = sm;
  __syncthreads();
  float tot = 0.f;
#pragma unroll
  for (int i = 0; i < 8; ++i) tot += red[i];
  float inv = 1.f / tot;
  e_s[tid] = a0 * inv; e_s[tid + 256] = a1 * inv;
  __syncthreads();
  // Phase 2: context[c] = sum_s alpha[s] * x[b,s,c]; coalesced across threads
  float acc0 = 0.f, acc1 = 0.f;
  const half_t* xb = xh + (size_t)b * S * C;
  for (int s = 0; s < S; ++s) {
    float a = e_s[s];
    acc0 += a * (float)xb[(size_t)s * C + tid];
    acc1 += a * (float)xb[(size_t)s * C + tid + 256];
  }
  ctx[(size_t)b * C + tid] = acc0;
  ctx[(size_t)b * C + tid + 256] = acc1;
  ctx_h[(size_t)b * C + tid] = (half_t)acc0;
  ctx_h[(size_t)b * C + tid + 256] = (half_t)acc1;
}

// Elementwise Keras-LSTM gates; writes h (f32 + f16) and hs_h[b,t,:] for W_gen.
__global__ void lstm_gates(const float* __restrict__ z, float* __restrict__ h,
                           float* __restrict__ c, half_t* __restrict__ h_h,
                           half_t* __restrict__ hs_h, int t) {
  int idx = blockIdx.x * blockDim.x + threadIdx.x;   // B*H = 65536
  int b = idx >> 9, j = idx & 511;
  const float* zb = z + (size_t)b * 2048;
  float zi = zb[j], zf = zb[512 + j], zg = zb[1024 + j], zo = zb[1536 + j];
  float ig = 1.f / (1.f + __expf(-zi));
  float fg = 1.f / (1.f + __expf(-zf));
  float gg = tanhf(zg);
  float og = 1.f / (1.f + __expf(-zo));
  float cn = fg * c[idx] + ig * gg;
  float hn = og * tanhf(cn);
  c[idx] = cn;
  h[idx] = hn;
  h_h[idx] = (half_t)hn;
  hs_h[((size_t)b * 25 + t) * 512 + j] = (half_t)hn;
}

__global__ void cvt_f32_f16(const float* __restrict__ src, half_t* __restrict__ dst, int n) {
  int i = blockIdx.x * blockDim.x + threadIdx.x;
  if (i < n) dst[i] = (half_t)src[i];
}

extern "C" void kernel_launch(void* const* d_in, const int* in_sizes, int n_in,
                              void* d_out, int out_size, void* d_ws, size_t ws_size,
                              hipStream_t stream) {
  (void)in_sizes; (void)n_in; (void)out_size; (void)ws_size;
  const float* x       = (const float*)d_in[0];
  const int*   text    = (const int*)  d_in[1];
  const float* W_i2h   = (const float*)d_in[2];
  const float* W_h2h   = (const float*)d_in[3];
  const float* b_h2h   = (const float*)d_in[4];
  const float* W_score = (const float*)d_in[5];
  const float* W_lstm  = (const float*)d_in[6];
  const float* U_lstm  = (const float*)d_in[7];
  const float* b_lstm  = (const float*)d_in[8];
  const float* W_gen   = (const float*)d_in[9];
  const float* b_gen   = (const float*)d_in[10];
  // d_in[11] = batch_max_length = 25 (hardcoded with the shapes)

  const int B = 128, S = 512, C = 512, H = 512, T = 25;

  char* ws = (char*)d_ws;
  size_t off = 0;
  auto alloc = [&](size_t bytes) -> char* {
    char* p = ws + off;
    off = (off + bytes + 255) & ~(size_t)255;
    return p;
  };
  half_t* Hproj_h = (half_t*)alloc((size_t)B * S * H * 2);   // 67 MB (L2-resident)
  half_t* x_h     = (half_t*)alloc((size_t)B * S * C * 2);   // 67 MB (L2-resident)
  half_t* Wi2h_h  = (half_t*)alloc((size_t)C * H * 2);
  half_t* Wh2h_h  = (half_t*)alloc((size_t)H * H * 2);
  half_t* Wl_h    = (half_t*)alloc((size_t)(C + 64) * 2048 * 2);
  half_t* U_h     = (half_t*)alloc((size_t)H * 2048 * 2);
  half_t* Wgen_h  = (half_t*)alloc((size_t)H * 64 * 2);
  float*  h       = (float*) alloc((size_t)B * H * 4);
  float*  c       = (float*) alloc((size_t)B * H * 4);
  half_t* h_h     = (half_t*)alloc((size_t)B * H * 2);
  float*  hp      = (float*) alloc((size_t)B * H * 4);
  float*  ctx     = (float*) alloc((size_t)B * C * 4);
  half_t* ctx_h   = (half_t*)alloc((size_t)B * C * 2);
  float*  z       = (float*) alloc((size_t)B * 2048 * 4);
  half_t* hs_h    = (half_t*)alloc((size_t)B * T * H * 2);

  // f32 -> f16 conversions (once per launch)
  auto cvt = [&](const float* s, half_t* d, int n) {
    cvt_f32_f16<<<(n + 255) / 256, 256, 0, stream>>>(s, d, n);
  };
  cvt(x,      x_h,    B * S * C);
  cvt(W_i2h,  Wi2h_h, C * H);
  cvt(W_h2h,  Wh2h_h, H * H);
  cvt(W_lstm, Wl_h,   (C + 64) * 2048);
  cvt(U_lstm, U_h,    H * 2048);
  cvt(W_gen,  Wgen_h, H * 64);

  hipMemsetAsync(h,   0, (size_t)B * H * 4, stream);
  hipMemsetAsync(c,   0, (size_t)B * H * 4, stream);
  hipMemsetAsync(h_h, 0, (size_t)B * H * 2, stream);

  // Loop-invariant: Hproj = x @ W_i2h  (65536 x 512 x 512), f16 output
  {
    int blocks = ((B * S) / 64) * (H / 32);
    gemm_f16<true, false><<<blocks, 256, 0, stream>>>(
        x_h, Wi2h_h, nullptr, nullptr, Hproj_h, B * S, H, C);
  }

  for (int t = 0; t < T; ++t) {
    // hp = h @ W_h2h + b_h2h   (128 x 512 x 512)
    {
      int blocks = (B / 64) * (H / 32);
      gemm_f16<false, true><<<blocks, 256, 0, stream>>>(
          h_h, Wh2h_h, b_h2h, hp, nullptr, B, H, H);
    }
    // attention + softmax + context
    attn_step<<<B, 256, 0, stream>>>(Hproj_h, x_h, hp, W_score, ctx, ctx_h);
    // z = ctx @ W_lstm[:C] + h @ U_lstm + b_lstm + onehot row
    {
      int blocks = (B / 64) * (2048 / 16);
      lstm_z<<<blocks, 128, 0, stream>>>(
          ctx_h, Wl_h, h_h, U_h, b_lstm, text, z, t);
    }
    // gates -> h, c, hs
    lstm_gates<<<(B * H) / 256, 256, 0, stream>>>(z, h, c, h_h, hs_h, t);
  }

  // probs = hs @ W_gen + b_gen  (3200 x 512 x 64) -> d_out (f32)
  {
    int blocks = ((B * T) / 64) * (64 / 32);
    gemm_f16<false, true><<<blocks, 256, 0, stream>>>(
        hs_h, Wgen_h, b_gen, (float*)d_out, nullptr, B * T, 64, H);
  }
}
